// GAT_62955630625070
// MI455X (gfx1250) — matrix operand order repaired and checked by reading, not compiled, not run on hardware
//
#include <hip/hip_runtime.h>

// ---------------------------------------------------------------------------
// GAT (2-layer, PyG GATConv semantics) for MI455X / gfx1250.
//
// Roofline:
//  * GEMM1 (50000x256 @ 256x256, 6.55 GFLOP) + GEMM2 (N pad 40->48) ->
//    v_wmma_f32_16x16x32_f16. Fragments now loaded as ds_load_b128 pairs
//    (LDS strides padded to 40 halves for 16B alignment + bank tiling).
//  * Edge softmax/aggregate: 850K edges x 8 heads, L2-resident float atomics
//    (192MB L2 >> ~100MB working set). Three flat passes.
// ---------------------------------------------------------------------------

typedef __attribute__((ext_vector_type(16))) _Float16 v16h;
typedef __attribute__((ext_vector_type(8)))  _Float16 v8h;
typedef __attribute__((ext_vector_type(8)))  float    v8f;

#define GAT_N       50000
#define GAT_E       800000LL
#define GAT_INF     256
#define GAT_HID     32
#define GAT_HEADS   8
#define GAT_F1      256           // HEADS*HID
#define GAT_NCLASS  40
#define GAT_N2PAD   48            // NCLASS padded to multiple of 16

#define ASTRIDE     40            // padded LDS row stride (halves), 80B
#define BSTRIDE     40

// ---- helpers ---------------------------------------------------------------

__device__ __forceinline__ unsigned f2ord(float x) {
  unsigned u = __float_as_uint(x);
  return (u & 0x80000000u) ? ~u : (u | 0x80000000u);
}
__device__ __forceinline__ float ord2f(unsigned u) {
  return (u & 0x80000000u) ? __uint_as_float(u & 0x7fffffffu)
                           : __uint_as_float(~u);
}
__device__ __forceinline__ float leaky(float x) { return x >= 0.f ? x : 0.2f * x; }

__device__ __forceinline__ void edge_sd(const long long* ei, long long E,
                                        long long e, int& s, int& d) {
  if (e < E) { s = (int)ei[e]; d = (int)ei[E + e]; }
  else       { s = d = (int)(e - E); }          // appended self loops
}

// ---- weight conversion f32 -> f16 (with zero column padding) ---------------

__global__ void cvt_w_f16(const float* __restrict__ W, _Float16* __restrict__ out,
                          int rows, int cols, int ocols) {
  int t = blockIdx.x * 256 + threadIdx.x;
  int total = rows * ocols;
  if (t >= total) return;
  int r = t / ocols, c = t % ocols;
  out[t] = (c < cols) ? (_Float16)W[(size_t)r * cols + c] : (_Float16)0.f;
}

// ---- WMMA GEMM: C[M,Npad] = A[M,K](f32, cvt->f16) * B[K,Npad](f16) ---------
// One block = 16 output rows; 8 waves each own N-tiles {w, w+8} (Npad<=256).
// A staged [16][32] (padded) row-major; B staged TRANSPOSED [Npad][32] so each
// lane's 16 K-values are contiguous -> fragments load as 2x ds_load_b128.

__global__ void __launch_bounds__(256) gemm_wmma_f16(
    const float* __restrict__ A, const _Float16* __restrict__ B,
    float* __restrict__ C, int M, int Npad, int K) {
  __shared__ _Float16 sA[16 * ASTRIDE];
  __shared__ _Float16 sBt[256 * BSTRIDE];
  const int tid  = threadIdx.x;
  const int lane = tid & 31;                                  // wave32
  const int wave = __builtin_amdgcn_readfirstlane(tid >> 5);  // scalar branch
  const int mbase  = blockIdx.x << 4;
  const int ntiles = Npad >> 4;
  const int t0 = wave, t1 = wave + 8;
  const bool do0 = (t0 < ntiles), do1 = (t1 < ntiles);        // scalar preds
  v8f acc0 = {}; v8f acc1 = {};

  // A 16x32 frag, contiguous k order used here (same for B): lane m=L&15; lanes 0-15 hold K 0..15,
  // lanes 16-31 hold K 16..31: element j -> K = kA + j
  const int mrow = lane & 15;
  const int kA   = (lane >> 4) << 4;      // +16 for upper half-wave
  // B 32x16 frag, same k order as A: N = lane&15; lanes 0-15 K=j, lanes 16-31 K=16+j
  const int kB   = lane & 16;
  const int nc0  = (t0 << 4) + (lane & 15);
  const int nc1  = (t1 << 4) + (lane & 15);

  for (int k0 = 0; k0 < K; k0 += 32) {
    __syncthreads();
    // stage A tile 16x32 (f32 -> f16), 2 elements/thread
    for (int i = tid; i < 16 * 32; i += 256) {
      int r = i >> 5, c = i & 31;
      int row = mbase + r;
      float v = (row < M) ? A[(size_t)row * K + (k0 + c)] : 0.f;
      sA[r * ASTRIDE + c] = (_Float16)v;
    }
    // stage B slab 32 x Npad, transposed into sBt[c][k]
    const _Float16* Bs = B + (size_t)k0 * Npad;
    for (int i = tid; i < 32 * Npad; i += 256) {
      int k = i / Npad, c = i - k * Npad;
      sBt[c * BSTRIDE + k] = Bs[i];
    }
    __syncthreads();

    union { v16h v; v8h g[2]; } a;
    const _Float16* ap = &sA[mrow * ASTRIDE + kA];
    a.g[0] = *(const v8h*)(ap);        // K = kA..kA+7
    a.g[1] = *(const v8h*)(ap + 8);    // K = kA+8..kA+15
    if (do0) {
      union { v16h v; v8h g[2]; } b;
      const _Float16* bp = &sBt[nc0 * BSTRIDE + kB];
      b.g[0] = *(const v8h*)(bp);
      b.g[1] = *(const v8h*)(bp + 8);
      acc0 = __builtin_amdgcn_wmma_f32_16x16x32_f16(false, a.v, false, b.v,
                                                    (short)0, acc0, false, false);
    }
    if (do1) {
      union { v16h v; v8h g[2]; } b;
      const _Float16* bp = &sBt[nc1 * BSTRIDE + kB];
      b.g[0] = *(const v8h*)(bp);
      b.g[1] = *(const v8h*)(bp + 8);
      acc1 = __builtin_amdgcn_wmma_f32_16x16x32_f16(false, a.v, false, b.v,
                                                    (short)0, acc1, false, false);
    }
  }

  // C/D layout: VGPR r -> M = r (lanes 0-15) or 8+r (lanes 16-31); N = lane&15
  const int rbase = mbase + ((lane >> 4) << 3);
  if (do0) {
#pragma unroll
    for (int r = 0; r < 8; ++r) {
      int row = rbase + r;
      if (row < M) C[(size_t)row * Npad + nc0] = acc0[r];
    }
  }
  if (do1) {
#pragma unroll
    for (int r = 0; r < 8; ++r) {
      int row = rbase + r;
      if (row < M) C[(size_t)row * Npad + nc1] = acc1[r];
    }
  }
}

// ---- attention scores: a_src/a_dst[n,h] = <h[n,h,:], att_{src,dst}[h,:]> ----

__global__ void att_scores(const float* __restrict__ h,
                           const float* __restrict__ as_w,
                           const float* __restrict__ ad_w,
                           float* __restrict__ asrc, float* __restrict__ adst,
                           int N, int H, int Cc, int stride) {
  long long t = (long long)blockIdx.x * 256 + threadIdx.x;
  if (t >= (long long)N * H) return;
  int n = (int)(t / H), hh = (int)(t % H);
  const float* row = h + (size_t)n * stride + hh * Cc;
  const float* aw  = as_w + hh * Cc;
  const float* dw  = ad_w + hh * Cc;
  float s = 0.f, d = 0.f;
  for (int c = 0; c < Cc; ++c) { float v = row[c]; s += v * aw[c]; d += v * dw[c]; }
  asrc[t] = s; adst[t] = d;
}

// ---- fill ------------------------------------------------------------------

__global__ void fill_u32(unsigned* __restrict__ p, unsigned v, long long n) {
  long long t = (long long)blockIdx.x * 256 + threadIdx.x;
  if (t < n) p[t] = v;
}

// ---- edge pass 1: segment max over dst (ordered-uint atomicMax) ------------

__global__ void edge_max_k(const long long* __restrict__ ei,
                           const float* __restrict__ asrc,
                           const float* __restrict__ adst,
                           unsigned* __restrict__ smax,
                           long long E, int N, int H) {
  long long t = (long long)blockIdx.x * 256 + threadIdx.x;
  long long total = (E + N) * (long long)H;
  if (t >= total) return;
  long long e = t / H; int hh = (int)(t % H);
  int s, d; edge_sd(ei, E, e, s, d);
  float lg = leaky(asrc[(size_t)s * H + hh] + adst[(size_t)d * H + hh]);
  atomicMax(&smax[(size_t)d * H + hh], f2ord(lg));
}

// ---- edge pass 2: denom = segment_sum(exp(logit - max)) --------------------

__global__ void edge_den_k(const long long* __restrict__ ei,
                           const float* __restrict__ asrc,
                           const float* __restrict__ adst,
                           const unsigned* __restrict__ smax,
                           float* __restrict__ den,
                           long long E, int N, int H) {
  long long t = (long long)blockIdx.x * 256 + threadIdx.x;
  long long total = (E + N) * (long long)H;
  if (t >= total) return;
  long long e = t / H; int hh = (int)(t % H);
  int s, d; edge_sd(ei, E, e, s, d);
  float lg = leaky(asrc[(size_t)s * H + hh] + adst[(size_t)d * H + hh]);
  float m  = ord2f(smax[(size_t)d * H + hh]);
  atomicAdd(&den[(size_t)d * H + hh], __expf(lg - m));
}

// ---- edge pass 3: agg[dst] += alpha * h[src] -------------------------------

__global__ void edge_agg_k(const long long* __restrict__ ei,
                           const float* __restrict__ asrc,
                           const float* __restrict__ adst,
                           const unsigned* __restrict__ smax,
                           const float* __restrict__ den,
                           const float* __restrict__ h,
                           float* __restrict__ agg,
                           long long E, int N, int H, int Cc,
                           int hstride, int astride) {
  long long t = (long long)blockIdx.x * 256 + threadIdx.x;
  long long total = (E + N) * (long long)H;
  if (t >= total) return;
  long long e = t / H; int hh = (int)(t % H);
  int s, d; edge_sd(ei, E, e, s, d);
  float lg = leaky(asrc[(size_t)s * H + hh] + adst[(size_t)d * H + hh]);
  float m  = ord2f(smax[(size_t)d * H + hh]);
  float alpha = __expf(lg - m) / (den[(size_t)d * H + hh] + 1e-16f);
  const float* hrow = h + (size_t)s * hstride + hh * Cc;
  float* arow = agg + (size_t)d * astride + hh * Cc;
  __builtin_prefetch(hrow, 0, 0);   // global_prefetch_b8
  for (int c = 0; c < Cc; ++c) atomicAdd(&arow[c], hrow[c] * alpha);
}

// ---- bias + ELU (in place, layer-1 output -> layer-2 input) ----------------

__global__ void bias_elu(float* __restrict__ h, const float* __restrict__ b,
                         long long total, int cols) {
  long long t = (long long)blockIdx.x * 256 + threadIdx.x;
  if (t >= total) return;
  int c = (int)(t % cols);
  float v = h[t] + b[c];
  h[t] = v > 0.f ? v : (__expf(v) - 1.f);
}

// ---- final: out = log_softmax(agg2 + b2) over 40 classes, 1 wave/node ------

__global__ void logsoftmax40(const float* __restrict__ agg,
                             const float* __restrict__ b,
                             float* __restrict__ out, int N) {
  int wave = threadIdx.x >> 5;
  int lane = threadIdx.x & 31;
  int n = blockIdx.x * 8 + wave;
  if (n >= N) return;
  int c0 = lane, c1 = lane + 32;
  float v0 = agg[(size_t)n * GAT_NCLASS + c0] + b[c0];          // c0 < 32 < 40
  float v1 = (c1 < GAT_NCLASS) ? agg[(size_t)n * GAT_NCLASS + c1] + b[c1]
                               : -3.4e38f;
  float m = fmaxf(v0, v1);
  for (int off = 16; off; off >>= 1) m = fmaxf(m, __shfl_xor(m, off, 32));
  float s = __expf(v0 - m) + ((c1 < GAT_NCLASS) ? __expf(v1 - m) : 0.f);
  for (int off = 16; off; off >>= 1) s += __shfl_xor(s, off, 32);
  float ls = __logf(s);
  out[(size_t)n * GAT_NCLASS + c0] = v0 - m - ls;
  if (c1 < GAT_NCLASS) out[(size_t)n * GAT_NCLASS + c1] = v1 - m - ls;
}

// ---------------------------------------------------------------------------

extern "C" void kernel_launch(void* const* d_in, const int* in_sizes, int n_in,
                              void* d_out, int out_size, void* d_ws, size_t ws_size,
                              hipStream_t stream) {
  (void)in_sizes; (void)n_in; (void)out_size; (void)ws_size;
  const float*     x    = (const float*)d_in[0];
  const long long* ei   = (const long long*)d_in[1];
  const float*     W1   = (const float*)d_in[2];
  const float*     as1  = (const float*)d_in[3];
  const float*     ad1  = (const float*)d_in[4];
  const float*     b1   = (const float*)d_in[5];
  const float*     W2   = (const float*)d_in[6];
  const float*     as2  = (const float*)d_in[7];
  const float*     ad2  = (const float*)d_in[8];
  const float*     b2   = (const float*)d_in[9];
  float*           out  = (float*)d_out;

  const int  N = GAT_N, H1 = GAT_HEADS, C1 = GAT_HID, F1 = GAT_F1;
  const long long E = GAT_E, Etot = E + N;

  // ---- workspace arena (offsets 256B-aligned; layer-2 aliases dead bufs) --
  char* w = (char*)d_ws;
  float*    h1    = (float*)(w + 0);             // [N,256] f32   51,200,000 B
  float*    agg1  = (float*)(w + 51200000);      // [N,256] f32   51,200,000 B
  float*    asrc1 = (float*)(w + 102400000);     // [N,8]          1,600,000 B
  float*    adst1 = (float*)(w + 104000000);
  unsigned* smax1 = (unsigned*)(w + 105600000);
  float*    den1  = (float*)(w + 107200000);
  _Float16* W1h   = (_Float16*)(w + 108800000);  // 256x256 f16      131,072 B
  _Float16* W2h   = (_Float16*)(w + 108931072);  // 256x48  f16       24,576 B
  // layer-2 aliases (h1 dead after edge pass 3 + ELU; agg1 becomes input):
  float*    g2    = (float*)(w + 0);             // [N,48] f32     9,600,000 B
  float*    agg2  = (float*)(w + 16000000);      // [N,40] f32     8,000,000 B
  float*    asrc2 = asrc1;  float* adst2 = adst1;
  unsigned* smax2 = smax1;  float* den2  = den1;

  const int TPB = 256;
  auto blk = [](long long n) { return (unsigned)((n + 255) / 256); };

  // 0) weights -> f16 (W2 zero-padded 40 -> 48 cols)
  cvt_w_f16<<<blk(256 * 256), TPB, 0, stream>>>(W1, W1h, 256, 256, 256);
  cvt_w_f16<<<blk(256 * GAT_N2PAD), TPB, 0, stream>>>(W2, W2h, 256, GAT_NCLASS, GAT_N2PAD);

  // ---------------- layer 1 ----------------
  gemm_wmma_f16<<<(N + 15) / 16, TPB, 0, stream>>>(x, W1h, h1, N, F1, GAT_INF);

  att_scores<<<blk((long long)N * H1), TPB, 0, stream>>>(h1, as1, ad1, asrc1, adst1,
                                                         N, H1, C1, F1);

  fill_u32<<<blk((long long)N * H1), TPB, 0, stream>>>(smax1, 0u, (long long)N * H1);
  fill_u32<<<blk((long long)N * H1), TPB, 0, stream>>>((unsigned*)den1, 0u, (long long)N * H1);
  fill_u32<<<blk((long long)N * F1), TPB, 0, stream>>>((unsigned*)agg1, 0u, (long long)N * F1);

  long long t1 = Etot * H1;
  edge_max_k<<<blk(t1), TPB, 0, stream>>>(ei, asrc1, adst1, smax1, E, N, H1);
  edge_den_k<<<blk(t1), TPB, 0, stream>>>(ei, asrc1, adst1, smax1, den1, E, N, H1);
  edge_agg_k<<<blk(t1), TPB, 0, stream>>>(ei, asrc1, adst1, smax1, den1,
                                          h1, agg1, E, N, H1, C1, F1, F1);

  bias_elu<<<blk((long long)N * F1), TPB, 0, stream>>>(agg1, b1, (long long)N * F1, F1);

  // ---------------- layer 2 (heads=1, C=40, padded to 48) ----------------
  gemm_wmma_f16<<<(N + 15) / 16, TPB, 0, stream>>>(agg1, W2h, g2, N, GAT_N2PAD, F1);

  att_scores<<<blk((long long)N), TPB, 0, stream>>>(g2, as2, ad2, asrc2, adst2,
                                                    N, 1, GAT_NCLASS, GAT_N2PAD);

  fill_u32<<<blk((long long)N), TPB, 0, stream>>>(smax2, 0u, (long long)N);
  fill_u32<<<blk((long long)N), TPB, 0, stream>>>((unsigned*)den2, 0u, (long long)N);
  fill_u32<<<blk((long long)N * GAT_NCLASS), TPB, 0, stream>>>((unsigned*)agg2, 0u,
                                                               (long long)N * GAT_NCLASS);

  edge_max_k<<<blk(Etot), TPB, 0, stream>>>(ei, asrc2, adst2, smax2, E, N, 1);
  edge_den_k<<<blk(Etot), TPB, 0, stream>>>(ei, asrc2, adst2, smax2, den2, E, N, 1);
  edge_agg_k<<<blk(Etot), TPB, 0, stream>>>(ei, asrc2, adst2, smax2, den2,
                                            g2, agg2, E, N, 1, GAT_NCLASS,
                                            GAT_N2PAD, GAT_NCLASS);

  logsoftmax40<<<(N + 7) / 8, TPB, 0, stream>>>(agg2, b2, out, N);
}
